// LinearConvAttention_27023934226682
// MI455X (gfx1250) — compile-verified, matrix operand order checked by reading, not run on hardware
//
#include <hip/hip_runtime.h>
#include <hip/hip_bf16.h>

// MI455X (gfx1250) linear-conv-attention.
// Memory-bound (~1.2 GB traffic @ 23.3 TB/s ~= 55us floor); all GEMM work
// on v_wmma_f32_16x16x32_bf16 so compute never hits the critical path.

#define NSP   (48 * 48 * 48)   // 110592 spatial positions
#define HS    48

typedef __attribute__((ext_vector_type(16))) __bf16 v16bf;
typedef __attribute__((ext_vector_type(8)))  float  v8f;

// 16-bit A/B matrix element map (ISA 7.12.2): half h of lane L covers
// K = (h<8 ? h : h+8) + 8*(L>=16)
__device__ __forceinline__ int kmap(int h, int hi) {
    return (h < 8 ? h : h + 8) + (hi ? 8 : 0);
}

__device__ __forceinline__ v8f wmma_bf16(v16bf a, v16bf b, v8f c) {
    return __builtin_amdgcn_wmma_f32_16x16x32_bf16(false, a, false, b, (short)0, c, false, false);
}

// ---------------------------------------------------------------------------
// Kernel 0: swizzle [Wq;Wk] (256x256 f32) into WMMA A-tile-ordered bf16,
// zero S[256] and kv[2*4*32*64].
// Wsw layout: [mt 0..15][kt 0..7][lane 0..31][16 halves], flat == tid.
// ---------------------------------------------------------------------------
__global__ void __launch_bounds__(256) k_prep(const float* __restrict__ Wq,
                                              const float* __restrict__ Wk,
                                              __bf16* __restrict__ Wsw,
                                              float* __restrict__ S,
                                              float* __restrict__ kv) {
    int tid = blockIdx.x * 256 + threadIdx.x;           // grid = 256 blocks -> 65536
    if (tid < 65536) {
        int hidx = tid & 15;
        int lane = (tid >> 4) & 31;
        int kt   = (tid >> 9) & 7;
        int mt   = tid >> 12;
        int row  = mt * 16 + (lane & 15);
        int col  = kt * 32 + kmap(hidx, lane >> 4);
        float v = (row < 128) ? Wq[row * 256 + col] : Wk[(row - 128) * 256 + col];
        Wsw[tid] = (__bf16)v;
    }
    if (tid < 16384) kv[tid] = 0.0f;
    if (tid < 256)   S[tid]  = 0.0f;
}

// ---------------------------------------------------------------------------
// Kernel 1: Y[256,16] = Wqk(256x256) @ x[b][:, n0..n0+15]  via 16 WMMAs/wave,
// then per-position softmax of q-rows (per head, 32 ch) and exp of k-rows
// with global S accumulation. q_soft and ek stored f32 [b][128][N].
// ---------------------------------------------------------------------------
__global__ void __launch_bounds__(256) k_qk(const float* __restrict__ x,
                                            const __bf16* __restrict__ Wsw,
                                            float* __restrict__ qsoft,
                                            float* __restrict__ ek,
                                            float* __restrict__ S) {
    __shared__ float lds[256 * 16];   // phase 1: x tile [c][16]; phase 2: Y [row][16]

    int b  = blockIdx.x / (NSP / 16);
    int n0 = (blockIdx.x % (NSP / 16)) * 16;
    int t  = threadIdx.x;
    const float* xb = x + (size_t)b * 256 * NSP;

    // load x tile: thread t = channel, 16 contiguous positions (coalesced)
    {
        const float4* p = (const float4*)(xb + (size_t)t * NSP + n0);
#pragma unroll
        for (int i = 0; i < 4; ++i) {
            float4 f = p[i];
            lds[t * 16 + i * 4 + 0] = f.x;
            lds[t * 16 + i * 4 + 1] = f.y;
            lds[t * 16 + i * 4 + 2] = f.z;
            lds[t * 16 + i * 4 + 3] = f.w;
        }
    }
    __syncthreads();

    int w = t >> 5, lane = t & 31, hi = lane >> 4, ln = lane & 15;
    v8f acc0 = {}, acc1 = {};
#pragma unroll
    for (int kt = 0; kt < 8; ++kt) {
        v16bf A0 = *(const v16bf*)(Wsw + (((2 * w + 0) * 8 + kt) * 32 + lane) * 16);
        v16bf A1 = *(const v16bf*)(Wsw + (((2 * w + 1) * 8 + kt) * 32 + lane) * 16);
        v16bf Bt;
#pragma unroll
        for (int hh = 0; hh < 16; ++hh)
            Bt[hh] = (__bf16)lds[(kt * 32 + kmap(hh, hi)) * 16 + ln];
        acc0 = wmma_bf16(A0, Bt, acc0);
        acc1 = wmma_bf16(A1, Bt, acc1);
    }
    __syncthreads();

    // spill Y (rows 32w..32w+31) back into LDS per C/D layout
#pragma unroll
    for (int r = 0; r < 8; ++r) {
        lds[(w * 32 +      r + 8 * hi) * 16 + ln] = acc0[r];
        lds[(w * 32 + 16 + r + 8 * hi) * 16 + ln] = acc1[r];
    }
    __syncthreads();

    // q: softmax over 32 channels per head, per column
    if (t < 64) {
        int h = t >> 4, col = t & 15;
        float m = -1e30f;
        for (int r = 0; r < 32; ++r) m = fmaxf(m, lds[(h * 32 + r) * 16 + col]);
        float s = 0.0f;
        for (int r = 0; r < 32; ++r) s += __expf(lds[(h * 32 + r) * 16 + col] - m);
        float inv = 1.0f / s;
        for (int r = 0; r < 32; ++r)
            qsoft[((size_t)b * 128 + h * 32 + r) * NSP + n0 + col] =
                __expf(lds[(h * 32 + r) * 16 + col] - m) * inv;
    }
    // k: ek = exp(k) (|k| <~ 1.6, safe without max-shift), accumulate S
    if (t < 128) {
        float s = 0.0f;
        float* ep = ek + ((size_t)b * 128 + t) * NSP + n0;
        for (int j = 0; j < 16; ++j) {
            float e = __expf(lds[(128 + t) * 16 + j]);
            ep[j] = e;
            s += e;
        }
        atomicAdd(&S[b * 128 + t], s);
    }
}

// ---------------------------------------------------------------------------
// Kernel 2: depthwise 3x3x3 conv (v never materialized globally) fused with
// kv[b][h][32][64] += ek-tile(16x32) x v-tile(32x16) via WMMA; atomic f32
// reduction of per-block partials. Block = (b, h-row, 4 w-rows), chunks of
// 32 d-positions (second chunk zero-padded via ek=0). Chunk loop is fully
// unrolled so all d-boundary predicates constant-fold; sliding-window
// prefetch (global_prefetch_b8) covers the next w-row.
// ---------------------------------------------------------------------------
__global__ void __launch_bounds__(256) k_convkv(const float* __restrict__ x,
                                                const float* __restrict__ Wv,
                                                const float* __restrict__ bvp,
                                                const float* __restrict__ ek,
                                                float* __restrict__ kv) {
    __shared__ __bf16 v_lds[32 * 264];   // [pos][c], padded stride
    __shared__ __bf16 e_lds[128 * 33];   // [row][pos], padded stride

    int idx = blockIdx.x;
    int b   = idx / (48 * 12);
    int rem = idx % (48 * 12);
    int hh  = rem / 12;
    int w0  = (rem % 12) * 4;

    int t = threadIdx.x;                        // t == channel for conv stage
    int w = t >> 5, lane = t & 31, hi = lane >> 4, ln = lane & 15;
    int head = w & 3, mt = w >> 2;              // 8 waves -> (head 0..3, mtile 0..1)

    v8f acc[4] = {};
    float wgt[27];
#pragma unroll
    for (int j = 0; j < 27; ++j) wgt[j] = Wv[t * 27 + j];
    float bias = bvp[t];
    const float* xb = x + ((size_t)b * 256 + t) * NSP;

    for (int ww = w0; ww < w0 + 4; ++ww) {
        // prefetch the next new w-row of the sliding window into cache
        if (ww + 2 < HS) {
#pragma unroll
            for (int dh = -1; dh <= 1; ++dh) {
                int h2 = hh + dh;
                if (h2 >= 0 && h2 < HS)
                    __builtin_prefetch(xb + (h2 * HS + ww + 2) * HS, 0, 0);
            }
        }
#pragma unroll
        for (int chunk = 0; chunk < 2; ++chunk) {
            const int d0 = chunk * 32;
            float vv[32];
#pragma unroll
            for (int i = 0; i < 32; ++i) vv[i] = bias;
#pragma unroll
            for (int dh = -1; dh <= 1; ++dh) {
#pragma unroll
                for (int dw = -1; dw <= 1; ++dw) {
                    int h2 = hh + dh, w2 = ww + dw;
                    bool ok = (h2 >= 0) && (h2 < HS) && (w2 >= 0) && (w2 < HS);
                    const float* p = xb + (h2 * HS + w2) * HS;
#pragma unroll
                    for (int i = 0; i < 32; ++i) {
#pragma unroll
                        for (int dz = 0; dz < 3; ++dz) {
                            const int dd = d0 + i - 1 + dz;   // compile-time
                            if (dd >= 0 && dd < HS)
                                if (ok)
                                    vv[i] += p[dd] * wgt[(dh + 1) * 9 + (dw + 1) * 3 + dz];
                        }
                    }
                }
            }
#pragma unroll
            for (int i = 0; i < 32; ++i) v_lds[i * 264 + t] = (__bf16)vv[i];

            if (t < 128) {
                int n0 = (hh * HS + ww) * HS + d0;
                const float* epp = ek + ((size_t)b * 128 + t) * NSP + n0;
#pragma unroll
                for (int i = 0; i < 32; ++i) {
                    float e = (chunk == 1 && i >= 16) ? 0.0f : epp[i];  // d-padding
                    e_lds[t * 33 + i] = (__bf16)e;
                }
            }
            __syncthreads();

            v16bf A;
            int arow = head * 32 + mt * 16 + ln;
#pragma unroll
            for (int hx = 0; hx < 16; ++hx)
                A[hx] = e_lds[arow * 33 + kmap(hx, hi)];
#pragma unroll
            for (int nt = 0; nt < 4; ++nt) {
                v16bf Bt;
#pragma unroll
                for (int hx = 0; hx < 16; ++hx)
                    Bt[hx] = v_lds[kmap(hx, hi) * 264 + head * 64 + nt * 16 + ln];
                acc[nt] = wmma_bf16(A, Bt, acc[nt]);
            }
            __syncthreads();
        }
    }
    float* kvb = kv + ((size_t)b * 4 + head) * 32 * 64;
#pragma unroll
    for (int nt = 0; nt < 4; ++nt)
#pragma unroll
        for (int r = 0; r < 8; ++r)
            atomicAdd(&kvb[(mt * 16 + r + 8 * hi) * 64 + nt * 16 + ln], acc[nt][r]);
}

// ---------------------------------------------------------------------------
// Kernel 3: khat = kv / S, stored bf16 in A-tile layout for k_out.
// Asw layout: [b][h][mt 0..3][lane][16]; A(m=vc, k=kc) = kv[kc][vc]/S[kc].
// ---------------------------------------------------------------------------
__global__ void __launch_bounds__(256) k_norm(const float* __restrict__ kv,
                                              const float* __restrict__ S,
                                              __bf16* __restrict__ Asw) {
    int tid = blockIdx.x * 256 + threadIdx.x;     // grid = 64 blocks -> 16384
    if (tid < 16384) {
        int hidx = tid & 15;
        int lane = (tid >> 4) & 31;
        int mt   = (tid >> 9) & 3;
        int h    = (tid >> 11) & 3;
        int b    = tid >> 13;
        int m = mt * 16 + (lane & 15);
        int k = kmap(hidx, lane >> 4);
        float v = kv[(((b * 4) + h) * 32 + k) * 64 + m] / S[b * 128 + h * 32 + k];
        Asw[tid] = (__bf16)v;
    }
}

// ---------------------------------------------------------------------------
// Kernel 4: out[b][h*64+vc][n] = (khat^T @ q)[vc][n] / (Z[n] + eps),
// Z = sum_k q (ksum == 1 after k-softmax). 4 WMMAs per wave (64x32 @ 32x16).
// ---------------------------------------------------------------------------
__global__ void __launch_bounds__(256) k_out(const float* __restrict__ qsoft,
                                             const __bf16* __restrict__ Asw,
                                             float* __restrict__ out) {
    __shared__ float q_lds[128 * 33];
    __shared__ float z_lds[128];

    int b  = blockIdx.x / (NSP / 32);
    int n0 = (blockIdx.x % (NSP / 32)) * 32;
    int t  = threadIdx.x;

    {
        int row = t >> 1, seg = (t & 1) * 16;
        const float* p = qsoft + ((size_t)b * 128 + row) * NSP + n0 + seg;
#pragma unroll
        for (int i = 0; i < 16; i += 4) {
            float4 f = *(const float4*)(p + i);
            q_lds[row * 33 + seg + i + 0] = f.x;
            q_lds[row * 33 + seg + i + 1] = f.y;
            q_lds[row * 33 + seg + i + 2] = f.z;
            q_lds[row * 33 + seg + i + 3] = f.w;
        }
    }
    __syncthreads();
    if (t < 128) {
        int h = t >> 5, pos = t & 31;
        float s = 0.0f;
        for (int kc = 0; kc < 32; ++kc) s += q_lds[(h * 32 + kc) * 33 + pos];
        z_lds[t] = s;
    }
    __syncthreads();

    int w = t >> 5, lane = t & 31, hi = lane >> 4, ln = lane & 15;
    int h = w & 3, pg = w >> 2;
    int n = pg * 16 + ln;

    v16bf Bt;
#pragma unroll
    for (int hx = 0; hx < 16; ++hx)
        Bt[hx] = (__bf16)q_lds[(h * 32 + kmap(hx, hi)) * 33 + n];

    float invz = 1.0f / (z_lds[h * 32 + n] + 1e-6f);
    const __bf16* Ab = Asw + ((size_t)(b * 4 + h) * 4) * 32 * 16;
#pragma unroll
    for (int mt = 0; mt < 4; ++mt) {
        v16bf A = *(const v16bf*)(Ab + (mt * 32 + lane) * 16);
        v8f acc = {};
        acc = wmma_bf16(A, Bt, acc);
#pragma unroll
        for (int r = 0; r < 8; ++r)
            out[((size_t)b * 256 + h * 64 + mt * 16 + r + 8 * hi) * NSP + n0 + n] =
                acc[r] * invz;
    }
}

// ---------------------------------------------------------------------------
extern "C" void kernel_launch(void* const* d_in, const int* in_sizes, int n_in,
                              void* d_out, int out_size, void* d_ws, size_t ws_size,
                              hipStream_t stream) {
    const float* x  = (const float*)d_in[0];   // [2,256,48,48,48]
    const float* Wq = (const float*)d_in[1];   // [128,256]
    const float* Wk = (const float*)d_in[2];   // [128,256]
    const float* Wv = (const float*)d_in[3];   // [256,1,3,3,3]
    const float* bv = (const float*)d_in[4];   // [256]
    float* out = (float*)d_out;

    // workspace carve (fp32 elements): qsoft | ek | S | kv | Asw(bf16) | Wsw(bf16)
    const size_t Q = (size_t)2 * 128 * NSP;    // 28,311,552
    float*  qsoft = (float*)d_ws;
    float*  ekp   = qsoft + Q;
    float*  Sp    = ekp + Q;
    float*  kvp   = Sp + 256;
    __bf16* Asw   = (__bf16*)(kvp + 16384);
    __bf16* Wsw   = Asw + 16384;

    k_prep  <<<256,                256, 0, stream>>>(Wq, Wk, Wsw, Sp, kvp);
    k_qk    <<<2 * (NSP / 16),     256, 0, stream>>>(x, Wsw, qsoft, ekp, Sp);
    k_convkv<<<2 * 48 * 12,        256, 0, stream>>>(x, Wv, bv, ekp, kvp);
    k_norm  <<<64,                 256, 0, stream>>>(kvp, Sp, Asw);
    k_out   <<<2 * (NSP / 32),     256, 0, stream>>>(qsoft, Asw, out);
}
